// Transformer_Attention_21260088115885
// MI455X (gfx1250) — matrix operand order repaired and checked by reading, not compiled, not run on hardware
//
#include <hip/hip_runtime.h>
#include <math.h>
#include <stdint.h>

// ---------------------------------------------------------------------------
// Types for CDNA5 WMMA (wave32, 16x16x32 bf16 -> f32 accum)
// ---------------------------------------------------------------------------
typedef __attribute__((ext_vector_type(16))) __bf16 v16bf;
typedef __attribute__((ext_vector_type(8)))  __bf16 bf16x8;
typedef __attribute__((ext_vector_type(4)))  __bf16 bf16x4;
typedef __attribute__((ext_vector_type(8)))  float  v8f;

// Round-to-nearest-even f32 -> bf16
__device__ __forceinline__ __bf16 f2bf(float f) {
    unsigned u = __builtin_bit_cast(unsigned, f);
    u += 0x7FFFu + ((u >> 16) & 1u);
    unsigned short h = (unsigned short)(u >> 16);
    return __builtin_bit_cast(__bf16, h);
}

// Combine two 16B LDS/global chunks into one WMMA operand register block.
__device__ __forceinline__ v16bf frag_join(const __bf16* p, int second_off) {
    bf16x8 lo = *(const bf16x8*)(p);
    bf16x8 hi = *(const bf16x8*)(p + second_off);
    return __builtin_shufflevector(lo, hi, 0,1,2,3,4,5,6,7,8,9,10,11,12,13,14,15);
}

// ---------------------------------------------------------------------------
// 16-byte global -> LDS copy. Prefer CDNA5 async-to-LDS (ASYNCcnt path);
// fall back to load + ds_store if the builtin is unavailable.
// The builtin takes int4 (GCC vector_size(16)) pointers in AS1 / AS3.
// ---------------------------------------------------------------------------
#if defined(__has_builtin)
#if __has_builtin(__builtin_amdgcn_global_load_async_to_lds_b128) && \
    __has_builtin(__builtin_amdgcn_s_wait_asynccnt)
#define HAVE_ASYNC_LDS 1
#endif
#endif

#ifdef HAVE_ASYNC_LDS
typedef int v4i_gcc __attribute__((__vector_size__(4 * sizeof(int))));
typedef __attribute__((address_space(1))) v4i_gcc as1_v4i;
typedef __attribute__((address_space(3))) v4i_gcc as3_v4i;
__device__ __forceinline__ void cp16(const __bf16* g, __bf16* l) {
    // Integer round-trips: global flat addr == as1 addr; flat LDS addr low 32
    // bits == LDS byte offset (ISA 10.2 aperture mapping).
    as1_v4i* gp = (as1_v4i*)(unsigned long long)(uintptr_t)g;
    as3_v4i* lp = (as3_v4i*)(unsigned)(uintptr_t)l;
    __builtin_amdgcn_global_load_async_to_lds_b128(gp, lp, 0, 0);
}
__device__ __forceinline__ void stage_wait() {
    __builtin_amdgcn_s_wait_asynccnt(0);
}
#else
__device__ __forceinline__ void cp16(const __bf16* g, __bf16* l) {
    *(bf16x8*)l = *(const bf16x8*)g;   // global_load_b128 + ds_store_b128
}
__device__ __forceinline__ void stage_wait() {}
#endif

// ---------------------------------------------------------------------------
// Element-wise f32 -> bf16 conversion (vectorized x4)
// ---------------------------------------------------------------------------
__global__ __launch_bounds__(256) void cvt_f32_bf16(const float* __restrict__ in,
                                                    __bf16* __restrict__ out,
                                                    long n4) {
    long i = (long)blockIdx.x * blockDim.x + threadIdx.x;
    long stride = (long)gridDim.x * blockDim.x;
    const float4* in4 = (const float4*)in;
    bf16x4* out4 = (bf16x4*)out;
    for (; i < n4; i += stride) {
        float4 v = in4[i];
        bf16x4 o;
        o.x = f2bf(v.x); o.y = f2bf(v.y); o.z = f2bf(v.z); o.w = f2bf(v.w);
        out4[i] = o;
    }
}

// ---------------------------------------------------------------------------
// Batched GEMM: C[b] = (A[b] @ W[b]^T) * scale      (LDS double-buffered)
//   A : [M,K] bf16 row-major, row stride = K
//   W : [N,K] bf16 row-major, row stride = K   (nn.Linear weight layout)
//   C : MODE 0 -> f32 row-major  [M,N], row stride ldc (scaled)
//       MODE 1 -> bf16 row-major [M,N], row stride ldc
//       MODE 2 -> bf16 TRANSPOSED [N,M], row stride ldc (= M span)
// Block = 256 threads (8 waves, 2x4). Block tile 128x256, wave tile 64x64,
// K staged by 32 through LDS (rows padded to 40 elems to avoid bank conflicts).
// Requires: M % 128 == 0, N % 256 == 0, K % 32 == 0.
// ---------------------------------------------------------------------------
#define LDS_ROW 40   // 32 payload + 8 pad (80B row stride, 16B-aligned chunks)

template <int MODE>
__global__ __launch_bounds__(256) void gemm_wmma_bf16(
    const __bf16* __restrict__ A, const __bf16* __restrict__ W,
    void* __restrict__ Cv, int K, int ldc,
    long sA, long sW, long sC, float scale)
{
    __shared__ __bf16 As[2][128 * LDS_ROW];   // 10 KB / stage
    __shared__ __bf16 Bs[2][256 * LDS_ROW];   // 20 KB / stage

    const int t    = threadIdx.x;
    const int lane = t & 31;
    const int wave = t >> 5;
    const int r    = lane & 15;   // row of A-tile / col of B-tile
    const int part = lane >> 4;   // lane-half selects K sub-range

    const int mw = (wave >> 2) * 64;          // wave tile origin within block
    const int nw = (wave & 3) * 64;
    const int m0 = blockIdx.x * 128 + mw;     // global tile origin
    const int n0 = blockIdx.y * 256 + nw;

    A += (long)blockIdx.z * sA + (long)blockIdx.x * 128 * K;
    W += (long)blockIdx.z * sW + (long)blockIdx.y * 256 * K;

    v8f acc[4][4];
    #pragma unroll
    for (int i = 0; i < 4; ++i)
        #pragma unroll
        for (int j = 0; j < 4; ++j)
            acc[i][j] = (v8f)(0.0f);

    // Cooperative K-slice copy: A 128x32 (512 16B chunks), B 256x32 (1024).
    auto stage = [&](int kk, int buf) {
        const __bf16* Ag = A + kk;
        const __bf16* Wg = W + kk;
        __bf16* Al = &As[buf][0];
        __bf16* Bl = &Bs[buf][0];
        #pragma unroll
        for (int i = 0; i < 2; ++i) {
            int c = t + i * 256, row = c >> 2, cb = c & 3;
            cp16(Ag + (long)row * K + cb * 8, Al + row * LDS_ROW + cb * 8);
        }
        #pragma unroll
        for (int i = 0; i < 4; ++i) {
            int c = t + i * 256, row = c >> 2, cb = c & 3;
            cp16(Wg + (long)row * K + cb * 8, Bl + row * LDS_ROW + cb * 8);
        }
    };

    stage(0, 0);
    stage_wait();
    __syncthreads();

    for (int kk = 0; kk < K; kk += 32) {
        const int cur = (kk >> 5) & 1;
        if (kk + 32 < K) stage(kk + 32, cur ^ 1);  // async prefetch next slice

        // Fragment reads from LDS (ISA 7.12.2 operand layouts)
        const __bf16* aS = &As[cur][(mw + r) * LDS_ROW + part * 8];
        const __bf16* bS = &Bs[cur][(nw + r) * LDS_ROW + part * 8];
        v16bf a[4], b[4];
        #pragma unroll
        for (int i = 0; i < 4; ++i)
            a[i] = frag_join(aS + i * 16 * LDS_ROW, 16);  // K 0-7 | 16-23 halves
        #pragma unroll
        for (int j = 0; j < 4; ++j)
            b[j] = frag_join(bS + j * 16 * LDS_ROW, 16);  // K 0-7 | 16-23 halves
        #pragma unroll
        for (int i = 0; i < 4; ++i)
            #pragma unroll
            for (int j = 0; j < 4; ++j)
                acc[i][j] = __builtin_amdgcn_wmma_f32_16x16x32_bf16(
                    false, a[i], false, b[j], (short)0, acc[i][j], false, false);

        stage_wait();        // next-slice async copies landed
        __syncthreads();     // all waves done reading 'cur', writes visible
    }

    // D layout: lane half 'part', col n = r; VGPR v -> row m = v + 8*part.
    if (MODE == 0) {
        float* C = (float*)Cv + (long)blockIdx.z * sC;
        #pragma unroll
        for (int i = 0; i < 4; ++i)
            #pragma unroll
            for (int j = 0; j < 4; ++j) {
                const long col = n0 + j * 16 + r;
                #pragma unroll
                for (int rr = 0; rr < 8; ++rr) {
                    long row = m0 + i * 16 + part * 8 + rr;
                    C[row * (long)ldc + col] = acc[i][j][rr] * scale;
                }
            }
    } else if (MODE == 1) {
        __bf16* C = (__bf16*)Cv + (long)blockIdx.z * sC;
        #pragma unroll
        for (int i = 0; i < 4; ++i)
            #pragma unroll
            for (int j = 0; j < 4; ++j) {
                const long col = n0 + j * 16 + r;
                #pragma unroll
                for (int rr = 0; rr < 8; ++rr) {
                    long row = m0 + i * 16 + part * 8 + rr;
                    C[row * (long)ldc + col] = f2bf(acc[i][j][rr]);
                }
            }
    } else { // MODE == 2: store C^T, contiguous along m -> one b128 per subtile
        __bf16* C = (__bf16*)Cv + (long)blockIdx.z * sC;
        #pragma unroll
        for (int i = 0; i < 4; ++i)
            #pragma unroll
            for (int j = 0; j < 4; ++j) {
                const long trow = n0 + j * 16 + r;          // transposed row = n
                bf16x8 pk;
                #pragma unroll
                for (int rr = 0; rr < 8; ++rr) pk[rr] = f2bf(acc[i][j][rr]);
                *(bf16x8*)(C + trow * (long)ldc + m0 + i * 16 + part * 8) = pk;
            }
    }
}

// ---------------------------------------------------------------------------
// Row softmax over 2048 f32 -> bf16. One 256-thread block per row.
// ---------------------------------------------------------------------------
__global__ __launch_bounds__(256) void softmax_rows(const float* __restrict__ X,
                                                    __bf16* __restrict__ Y) {
    const int NC = 2048;
    const long row = blockIdx.x;
    const int t = threadIdx.x, lane = t & 31, wave = t >> 5;

    const float4* xv = (const float4*)(X + row * (long)NC);
    float4 v0 = xv[t];
    float4 v1 = xv[t + 256];

    float m = fmaxf(fmaxf(fmaxf(v0.x, v0.y), fmaxf(v0.z, v0.w)),
                    fmaxf(fmaxf(v1.x, v1.y), fmaxf(v1.z, v1.w)));
    #pragma unroll
    for (int off = 16; off; off >>= 1) m = fmaxf(m, __shfl_xor(m, off, 32));

    __shared__ float smax[8], ssum[8];
    if (lane == 0) smax[wave] = m;
    __syncthreads();
    m = smax[0];
    #pragma unroll
    for (int i = 1; i < 8; ++i) m = fmaxf(m, smax[i]);

    float e[8];
    e[0] = __expf(v0.x - m); e[1] = __expf(v0.y - m);
    e[2] = __expf(v0.z - m); e[3] = __expf(v0.w - m);
    e[4] = __expf(v1.x - m); e[5] = __expf(v1.y - m);
    e[6] = __expf(v1.z - m); e[7] = __expf(v1.w - m);
    float s = e[0] + e[1] + e[2] + e[3] + e[4] + e[5] + e[6] + e[7];
    #pragma unroll
    for (int off = 16; off; off >>= 1) s += __shfl_xor(s, off, 32);
    if (lane == 0) ssum[wave] = s;
    __syncthreads();
    s = 0.f;
    #pragma unroll
    for (int i = 0; i < 8; ++i) s += ssum[i];
    const float inv = 1.0f / s;

    bf16x4* yv = (bf16x4*)(Y + row * (long)NC);
    bf16x4 o0, o1;
    o0.x = f2bf(e[0] * inv); o0.y = f2bf(e[1] * inv);
    o0.z = f2bf(e[2] * inv); o0.w = f2bf(e[3] * inv);
    o1.x = f2bf(e[4] * inv); o1.y = f2bf(e[5] * inv);
    o1.z = f2bf(e[6] * inv); o1.w = f2bf(e[7] * inv);
    yv[t] = o0;
    yv[t + 256] = o1;
}

// ---------------------------------------------------------------------------
// Host-side orchestration
// ---------------------------------------------------------------------------
extern "C" void kernel_launch(void* const* d_in, const int* in_sizes, int n_in,
                              void* d_out, int out_size, void* d_ws, size_t ws_size,
                              hipStream_t stream) {
    const int B = 4, S = 2048, D = 1024;
    const long BS = (long)B * S;              // 8192
    const long XN = BS * D;                   // 8,388,608
    const long WN = (long)D * D;              // 1,048,576
    const long AN = (long)B * S * S;          // 16,777,216
    const float scale = 0.04419417382415922f; // (512)^-0.5

    const float* x  = (const float*)d_in[0];
    const float* Wq = (const float*)d_in[1];
    const float* Wk = (const float*)d_in[2];
    const float* Wv = (const float*)d_in[3];
    const float* Wo = (const float*)d_in[4];

    float* out_f32 = (float*)d_out;           // [B,S,D]
    float* weight  = out_f32 + XN;            // [B,S,S] (scaled pre-softmax)

    // Workspace carve-up (bf16 buffers)
    char* ws = (char*)d_ws;
    size_t off = 0;
    __bf16* xb  = (__bf16*)(ws + off); off += (size_t)XN * 2;
    __bf16* wqb = (__bf16*)(ws + off); off += (size_t)WN * 2;
    __bf16* wkb = (__bf16*)(ws + off); off += (size_t)WN * 2;
    __bf16* wvb = (__bf16*)(ws + off); off += (size_t)WN * 2;
    __bf16* wob = (__bf16*)(ws + off); off += (size_t)WN * 2;
    __bf16* qb  = (__bf16*)(ws + off); off += (size_t)XN * 2;
    __bf16* kb  = (__bf16*)(ws + off); off += (size_t)XN * 2;
    __bf16* vTb = (__bf16*)(ws + off); off += (size_t)XN * 2;   // [B][D][S]
    __bf16* ctx = (__bf16*)(ws + off); off += (size_t)XN * 2;   // [B][S][D]
    __bf16* atb = (__bf16*)(ws + off); off += (size_t)AN * 2;   // [B][S][S]

    // 1) Convert inputs to bf16
    cvt_f32_bf16<<<4096, 256, 0, stream>>>(x,  xb,  XN / 4);
    cvt_f32_bf16<<<1024, 256, 0, stream>>>(Wq, wqb, WN / 4);
    cvt_f32_bf16<<<1024, 256, 0, stream>>>(Wk, wkb, WN / 4);
    cvt_f32_bf16<<<1024, 256, 0, stream>>>(Wv, wvb, WN / 4);
    cvt_f32_bf16<<<1024, 256, 0, stream>>>(Wo, wob, WN / 4);

    const long sX = (long)S * D;   // per-batch stride of [S,D]
    const long sW2= (long)S * S;   // per-batch stride of [S,S]

    // 2) Projections: q, k (bf16 row-major), v stored transposed [D,S]
    dim3 gP(S / 128, D / 256, B);
    gemm_wmma_bf16<1><<<gP, 256, 0, stream>>>(xb, wqb, qb,  D, D, sX, 0, sX, 1.f);
    gemm_wmma_bf16<1><<<gP, 256, 0, stream>>>(xb, wkb, kb,  D, D, sX, 0, sX, 1.f);
    gemm_wmma_bf16<2><<<gP, 256, 0, stream>>>(xb, wvb, vTb, D, S, sX, 0, sX, 1.f);

    // 3) dist = (q @ k^T) * scale  -> f32 weight output region
    dim3 gD(S / 128, S / 256, B);
    gemm_wmma_bf16<0><<<gD, 256, 0, stream>>>(qb, kb, weight, D, S, sX, sX, sW2, scale);

    // 4) attn = softmax(dist) -> bf16
    softmax_rows<<<B * S, 256, 0, stream>>>(weight, atb);

    // 5) ctx = attn @ v  (v^T is [D,S] row-major with row stride S = K)
    dim3 gC(S / 128, D / 256, B);
    gemm_wmma_bf16<1><<<gC, 256, 0, stream>>>(atb, vTb, ctx, S, D, sW2, sX, sX, 1.f);

    // 6) out = ctx @ Wo^T -> f32
    gemm_wmma_bf16<0><<<gC, 256, 0, stream>>>(ctx, wob, out_f32, D, D, sX, 0, sX, 1.f);
}
